// QuantizedLinear_79955111182858
// MI455X (gfx1250) — matrix operand order repaired and checked by reading, not compiled, not run on hardware
//
#include <hip/hip_runtime.h>

// Ternary-quantized linear: out[64,8192] = x[64,8192] @ q^T[8192,8192] + bias
// Memory-bound on the 256MB fp32 weight stream @ 23.3 TB/s (~11.5us).
//  Kernel 1 (prepass): split fp32 x once into bf16 hi+lo arrays in d_ws (2 MB).
//  Kernel 2 (GEMM):  4-deep LDS ring filled by GLOBAL_LOAD_ASYNC_TO_LDS (ASYNCcnt),
//                    ~72KB in flight per WGP to cover HBM latency; no VGPR staging.
//                    Weight arrives raw f32 in LDS; a tiny LDS->LDS stage ternarizes
//                    it to exact bf16 {-1,0,+1}. 2 chained v_wmma_f32_16x16x32_bf16
//                    (hi,lo) per k-step, f32 acc. Steady-state loop is branchless:
//                    unconditional s_wait_asynccnt 12 + unconditional refill; a
//                    3-iteration drain loop handles the pipeline tail.
//                    Epilogue: out = alpha[n]*acc + bias[n] in f32.

typedef __attribute__((ext_vector_type(16))) __bf16 v16bf;
typedef __attribute__((ext_vector_type(8)))  float  v8f;
typedef __attribute__((__vector_size__(16))) int    v4i_t;
typedef __attribute__((address_space(1))) v4i_t*    gas_v4i;  // global (AS1)
typedef __attribute__((address_space(3))) v4i_t*    las_v4i;  // LDS (AS3)

#define B_M    64
#define K_I    8192
#define N_O    8192
#define KT     64      // K per chunk (2 wmma k-steps)
#define NT     32      // N tile per block
#define LDSK   72      // padded ushort row stride (144B, 16B-aligned, conflict-free)
#define NCHUNK (K_I / KT)
#define DEPTH  4       // LDS ring depth (3 chunks in flight)

#if __has_builtin(__builtin_amdgcn_global_load_async_to_lds_b128)
#define HAVE_ASYNC 1
#else
#define HAVE_ASYNC 0
#endif

#if HAVE_ASYNC
#if __has_builtin(__builtin_amdgcn_s_wait_asynccnt)
#define WAIT_ASYNC(n) __builtin_amdgcn_s_wait_asynccnt(n)
#else
#define WAIT_ASYNC(n) asm volatile("s_wait_asynccnt " #n ::: "memory")
#endif
#else
#define WAIT_ASYNC(n) ((void)0)
#endif

// 16B global -> LDS copy: async (ASYNCcnt) on gfx1250, sync fallback otherwise.
__device__ __forceinline__ void copy16(const void* g, void* l) {
#if HAVE_ASYNC
  __builtin_amdgcn_global_load_async_to_lds_b128((gas_v4i)g, (las_v4i)l, 0, 0);
#else
  *(uint4*)l = *(const uint4*)g;
#endif
}

static __device__ __forceinline__ unsigned short bf16_rne(float f) {
  unsigned u = __builtin_bit_cast(unsigned, f);
  u += 0x7FFFu + ((u >> 16) & 1u);
  return (unsigned short)(u >> 16);
}

// ---------- prepass: x (f32) -> x_hi + x_lo (bf16), ~16-bit-significand split ----------
__global__ __launch_bounds__(256) void xsplit_kernel(
    const float* __restrict__ x,
    unsigned short* __restrict__ xhi, unsigned short* __restrict__ xlo) {
  const int i = blockIdx.x * 256 + threadIdx.x;
  float4 v = ((const float4*)x)[i];
  ushort4 hh, ll;
  hh.x = bf16_rne(v.x); ll.x = bf16_rne(v.x - __builtin_bit_cast(float, (unsigned)hh.x << 16));
  hh.y = bf16_rne(v.y); ll.y = bf16_rne(v.y - __builtin_bit_cast(float, (unsigned)hh.y << 16));
  hh.z = bf16_rne(v.z); ll.z = bf16_rne(v.z - __builtin_bit_cast(float, (unsigned)hh.z << 16));
  hh.w = bf16_rne(v.w); ll.w = bf16_rne(v.w - __builtin_bit_cast(float, (unsigned)hh.w << 16));
  ((ushort4*)xhi)[i] = hh;
  ((ushort4*)xlo)[i] = ll;
}

// ---------- main GEMM ----------
__global__ __launch_bounds__(256) void ternary_linear_wmma(
    const unsigned short* __restrict__ xhi, const unsigned short* __restrict__ xlo,
    const float* __restrict__ w,
    const float* __restrict__ alpha, const float* __restrict__ bias,
    float* __restrict__ out) {

  __shared__ __align__(16) unsigned short lds_xhi[DEPTH][B_M * LDSK]; // 36 KB
  __shared__ __align__(16) unsigned short lds_xlo[DEPTH][B_M * LDSK]; // 36 KB
  __shared__ __align__(16) float          lds_wraw[DEPTH][NT * KT];   // 32 KB
  __shared__ __align__(16) unsigned short lds_wbf[NT * LDSK];         // 4.5 KB

  const int tid  = threadIdx.x;
  const int n0   = blockIdx.x * NT;
  const int lane = tid & 31;
  const int wv   = tid >> 5;
  const int m0   = (wv & 3) * 16;
  const int ntl  = (wv >> 2) * 16;
  // fragment addressing per CDNA5 16-bit WMMA VGPR layout
  const int arow = m0  + (lane & 15);
  const int aK0  = (lane < 16) ? 0 : 8;
  const int brow = ntl + (lane & 15);
  const int bK0  = (lane < 16) ? 0 : 8;

  float thrj[2];
#pragma unroll
  for (int j = 0; j < 2; ++j) {
    int r = (tid + j * 256) >> 4;
    thrj[j] = 0.5f * (alpha[n0 + r] + 1e-8f);
  }

  // issue one chunk's copies into ring slot: 6x 16B async per thread
  auto fill = [&](int c) {
    const int k0   = c * KT;
    const int slot = c & (DEPTH - 1);
#pragma unroll
    for (int j = 0; j < 2; ++j) {              // x hi/lo: 512 x 16B each
      int idx = tid + j * 256, r = idx >> 3, c8 = idx & 7;
      int go = r * K_I + k0 + c8 * 8;
      int lo = r * LDSK + c8 * 8;
      copy16(xhi + go, &lds_xhi[slot][lo]);
      copy16(xlo + go, &lds_xlo[slot][lo]);
    }
#pragma unroll
    for (int j = 0; j < 2; ++j) {              // w raw f32: 512 x 16B
      int idx = tid + j * 256, r = idx >> 4, c4 = idx & 15;
      copy16(w + (n0 + r) * K_I + k0 + c4 * 4,
             &lds_wraw[slot][r * KT + c4 * 4]);
    }
  };

  // LDS->LDS ternarization of the weight chunk (branchless, exact bf16 {-1,0,+1})
  auto quantize = [&](int c) {
    const int slot = c & (DEPTH - 1);
#pragma unroll
    for (int j = 0; j < 2; ++j) {
      int idx = tid + j * 256, r = idx >> 4, c4 = idx & 15;
      float4 v = *(const float4*)(&lds_wraw[slot][r * KT + c4 * 4]);
      const float th = thrj[j];
      unsigned bx = __builtin_bit_cast(unsigned, v.x);
      unsigned by = __builtin_bit_cast(unsigned, v.y);
      unsigned bz = __builtin_bit_cast(unsigned, v.z);
      unsigned bw = __builtin_bit_cast(unsigned, v.w);
      ushort4 t;
      t.x = (__builtin_fabsf(v.x) > th) ? (unsigned short)(0x3F80u | ((bx >> 16) & 0x8000u)) : (unsigned short)0;
      t.y = (__builtin_fabsf(v.y) > th) ? (unsigned short)(0x3F80u | ((by >> 16) & 0x8000u)) : (unsigned short)0;
      t.z = (__builtin_fabsf(v.z) > th) ? (unsigned short)(0x3F80u | ((bz >> 16) & 0x8000u)) : (unsigned short)0;
      t.w = (__builtin_fabsf(v.w) > th) ? (unsigned short)(0x3F80u | ((bw >> 16) & 0x8000u)) : (unsigned short)0;
      *(ushort4*)(&lds_wbf[r * LDSK + c4 * 4]) = t;
    }
  };

  v8f acc = {};

  auto compute = [&](int c) {
    const int slot = c & (DEPTH - 1);
    const unsigned short* xh = &lds_xhi[slot][arow * LDSK];
    const unsigned short* xl = &lds_xlo[slot][arow * LDSK];
    const unsigned short* bt = &lds_wbf[brow * LDSK];
#pragma unroll
    for (int s = 0; s < 2; ++s) {
      union { v16bf v; uint4 q[2]; } ub, ua, ul;
      const unsigned short* pb = bt + s * 32 + bK0;
      ub.q[0] = *(const uint4*)(pb);
      ub.q[1] = *(const uint4*)(pb + 16);
      const unsigned short* ph = xh + s * 32;
      ua.q[0] = *(const uint4*)(ph + aK0);
      ua.q[1] = *(const uint4*)(ph + aK0 + 16);
      acc = __builtin_amdgcn_wmma_f32_16x16x32_bf16(
          false, ua.v, false, ub.v, (short)0, acc, false, false);
      const unsigned short* pl = xl + s * 32;
      ul.q[0] = *(const uint4*)(pl + aK0);
      ul.q[1] = *(const uint4*)(pl + aK0 + 16);
      acc = __builtin_amdgcn_wmma_f32_16x16x32_bf16(
          false, ul.v, false, ub.v, (short)0, acc, false, false);
    }
  };

  // --- software pipeline: DEPTH-1 chunks in flight via ASYNCcnt ---
#pragma unroll
  for (int d = 0; d < DEPTH - 1; ++d) fill(d);   // chunks 0..2

  int c = 0;
  // steady state, branchless: chunk c done when <= 6*(DEPTH-2)=12 ops outstanding;
  // refills chunks 3..127 exactly once.
#pragma unroll 1
  for (; c < NCHUNK - DEPTH + 1; ++c) {
    WAIT_ASYNC(12);
    __syncthreads();                            // tile c visible to all waves
    fill(c + DEPTH - 1);                        // refill freed slot (unconditional)
    quantize(c);                                // w f32 -> ternary bf16 in LDS
    __syncthreads();
    compute(c);                                 // 4x v_wmma from LDS
  }
  // drain: last 3 chunks, copies long since issued -> conservative full wait
#pragma unroll 1
  for (; c < NCHUNK; ++c) {
    WAIT_ASYNC(0);
    __syncthreads();
    quantize(c);
    __syncthreads();
    compute(c);
  }

  // --- epilogue: out = alpha[n]*acc + bias[n], per CDNA5 C/D layout ---
  const int nn = n0 + ntl + (lane & 15);
  const int mb = m0 + ((lane >> 4) << 3);
  const float as = alpha[nn];
  const float bs = bias[nn];
#pragma unroll
  for (int r = 0; r < 8; ++r)
    out[(mb + r) * N_O + nn] = acc[r] * as + bs;
}

extern "C" void kernel_launch(void* const* d_in, const int* in_sizes, int n_in,
                              void* d_out, int out_size, void* d_ws, size_t ws_size,
                              hipStream_t stream) {
  (void)in_sizes; (void)n_in; (void)out_size; (void)ws_size;
  const float* x     = (const float*)d_in[0];
  const float* w     = (const float*)d_in[1];
  const float* alpha = (const float*)d_in[2];
  const float* bias  = (const float*)d_in[3];
  float* out = (float*)d_out;

  unsigned short* xhi = (unsigned short*)d_ws;      // 1 MB
  unsigned short* xlo = xhi + (size_t)B_M * K_I;    // 1 MB

  xsplit_kernel<<<dim3((B_M * K_I / 4) / 256), 256, 0, stream>>>(x, xhi, xlo);
  ternary_linear_wmma<<<dim3(N_O / NT), 256, 0, stream>>>(xhi, xlo, w, alpha, bias, out);
}
